// GPT2Attention_80436147519755
// MI455X (gfx1250) — compile-verified, matrix operand order checked by reading, not run on hardware
//
#include <hip/hip_runtime.h>

// ---------------------------------------------------------------------------
// GPT-2 attention on MI455X (gfx1250): bf16 WMMA everywhere + async global->LDS
// ---------------------------------------------------------------------------
#define BB   4
#define SS   2048
#define HH   2048
#define NHD  16
#define HD   128
#define MTOT (BB * SS)   // 8192 rows

typedef __bf16 bf16;
typedef __bf16 v16bf __attribute__((ext_vector_type(16)));
typedef float  v8f   __attribute__((ext_vector_type(8)));
typedef int    v4i   __attribute__((vector_size(16)));   // matches builtin proto

union Frag  { v16bf v; uint4 u[2]; };
union Pack8 { bf16 h[8]; uint4 u; };

// Load one 16-lane-row / 32-K bf16 fragment (A-layout: lane&15 = row,
// lane>>4 = ksel; elements {k..k+7} and {k+16..k+23}) = two 16B loads.
__device__ __forceinline__ v16bf ld_frag(const bf16* p) {
    Frag f;
    f.u[0] = *(const uint4*)(p);
    f.u[1] = *(const uint4*)(p + 16);
    return f.v;
}

__device__ __forceinline__ v8f wmma_bf16(v16bf a, v16bf b, v8f c) {
    return __builtin_amdgcn_wmma_f32_16x16x32_bf16(
        /*neg_a=*/false, a, /*neg_b=*/false, b,
        /*c_mod=*/(short)0, c, /*reuse_a=*/false, /*reuse_b=*/false);
}

__device__ __forceinline__ uint4 cvt8_f32_bf16(const float* p) {
    float4 x = *(const float4*)p;
    float4 y = *(const float4*)(p + 4);
    Pack8 r;
    r.h[0] = (bf16)x.x; r.h[1] = (bf16)x.y; r.h[2] = (bf16)x.z; r.h[3] = (bf16)x.w;
    r.h[4] = (bf16)y.x; r.h[5] = (bf16)y.y; r.h[6] = (bf16)y.z; r.h[7] = (bf16)y.w;
    return r.u;
}

// ---- CDNA5 async global->LDS copy (ASYNCcnt-tracked, no VGPR staging) ------
__device__ __forceinline__ void async_copy_b128(bf16* lds_dst, const bf16* gsrc) {
#if __has_builtin(__builtin_amdgcn_global_load_async_to_lds_b128)
    __builtin_amdgcn_global_load_async_to_lds_b128(
        (__attribute__((address_space(1))) v4i*)(void*)gsrc,
        (__attribute__((address_space(3))) v4i*)(void*)lds_dst,
        0, 0);
#else
    unsigned off = (unsigned)(unsigned long long)lds_dst;  // low 32b = LDS offset
    asm volatile("global_load_async_to_lds_b128 %0, %1, off"
                 :: "v"(off), "v"(gsrc) : "memory");
#endif
}

__device__ __forceinline__ void wait_async0() {
#if __has_builtin(__builtin_amdgcn_s_wait_asynccnt)
    __builtin_amdgcn_s_wait_asynccnt(0);
#else
    asm volatile("s_wait_asynccnt 0" ::: "memory");
#endif
}

// ---------------------------------------------------------------------------
// GEMM: Out[m,n] = sum_k A[m,k] * W[n,k] + bias[n]   (i.e. A @ W^T + b)
// Block tile 64(M) x 256(N), 256 threads = 8 waves, wave tile 32x64, Ktile 32.
// f32 operands are converted to bf16 in VALU during LDS staging.
// OUT_MODE: 0 = bf16 row-major   1 = bf16 Vt[b,h,d,s]   2 = f32 row-major
// ---------------------------------------------------------------------------
template <int OUT_MODE, bool A_BF16>
__global__ __launch_bounds__(256) void gemm_wmma_kernel(
    const void* __restrict__ Av, const float* __restrict__ W,
    const float* __restrict__ bias, void* __restrict__ Out)
{
    __shared__ bf16 sA[64 * 40];     // 80B row stride (pad) -> conflict-friendly
    __shared__ bf16 sB[256 * 40];

    const int tid   = threadIdx.x;
    const int lane  = tid & 31;
    const int wave  = tid >> 5;
    const int laneN = lane & 15;
    const int ksel  = lane >> 4;
    const int wm    = wave & 1;       // 2 waves across M (2 x 32 = 64)
    const int wn    = wave >> 1;      // 4 waves across N (4 x 64 = 256)
    const int rowBase = blockIdx.x * 64;
    const int colBase = blockIdx.y * 256;

    const int ra = tid >> 2;          // A staging row 0..63
    const int qa = tid & 3;           // A staging 8-elem chunk

    const float* Af = (const float*)Av;
    const bf16*  Ah = (const bf16*)Av;

    v8f c[2][4];
    #pragma unroll
    for (int i = 0; i < 2; i++)
        #pragma unroll
        for (int j = 0; j < 4; j++)
            c[i][j] = (v8f){0.f, 0.f, 0.f, 0.f, 0.f, 0.f, 0.f, 0.f};

    for (int kt = 0; kt < HH; kt += 32) {
        __syncthreads();
        // ---- stage A tile [64 x 32] ----
        {
            uint4 pk;
            if (A_BF16) {
                pk = *(const uint4*)(Ah + (size_t)(rowBase + ra) * HH + kt + qa * 8);
            } else {
                pk = cvt8_f32_bf16(Af + (size_t)(rowBase + ra) * HH + kt + qa * 8);
            }
            *(uint4*)(sA + ra * 40 + qa * 8) = pk;
        }
        // ---- stage B tile [256 x 32] from W rows (N-major, K contiguous) ----
        #pragma unroll
        for (int i = 0; i < 4; i++) {
            const int cch = tid + i * 256;          // chunk 0..1023
            const int br  = cch >> 2;               // row 0..255
            const int bq  = cch & 3;
            uint4 pk = cvt8_f32_bf16(W + (size_t)(colBase + br) * HH + kt + bq * 8);
            *(uint4*)(sB + br * 40 + bq * 8) = pk;
        }
        if (kt + 32 < HH) {
            if (!A_BF16)
                __builtin_prefetch(Af + (size_t)(rowBase + ra) * HH + kt + 32, 0, 1);
            __builtin_prefetch(W + (size_t)(colBase + ra) * HH + kt + 32, 0, 1);
        }
        __syncthreads();

        // ---- 8 WMMAs: 32x64 wave tile ----
        v16bf a0 = ld_frag(sA + (wm * 32 + 0  + laneN) * 40 + ksel * 8);
        v16bf a1 = ld_frag(sA + (wm * 32 + 16 + laneN) * 40 + ksel * 8);
        #pragma unroll
        for (int j = 0; j < 4; j++) {
            v16bf bj = ld_frag(sB + (wn * 64 + j * 16 + laneN) * 40 + ksel * 8);
            c[0][j] = wmma_bf16(a0, bj, c[0][j]);
            c[1][j] = wmma_bf16(a1, bj, c[1][j]);
        }
    }

    // ---- epilogue: C frag row m = g + 8*(lane>=16), col n = lane&15 ----
    const int mrow = 8 * ksel;
    #pragma unroll
    for (int fn = 0; fn < 4; fn++) {
        const int gn = colBase + wn * 64 + fn * 16 + laneN;
        const float bv = bias[gn];
        #pragma unroll
        for (int fm = 0; fm < 2; fm++) {
            const int gmBase = rowBase + wm * 32 + fm * 16 + mrow;
            #pragma unroll
            for (int g = 0; g < 8; g++) {
                const int gm = gmBase + g;
                const float val = c[fm][fn][g] + bv;
                if (OUT_MODE == 0) {
                    ((bf16*)Out)[(size_t)gm * HH + gn] = (bf16)val;
                } else if (OUT_MODE == 1) {
                    const int b = gm >> 11, s = gm & (SS - 1);
                    const int head = gn >> 7, d = gn & (HD - 1);
                    ((bf16*)Out)[((size_t)(b * NHD + head) * HD + d) * SS + s] = (bf16)val;
                } else {
                    ((float*)Out)[(size_t)gm * HH + gn] = val;
                }
            }
        }
    }
}

// ---------------------------------------------------------------------------
// Causal flash attention: grid (S/128, B*NH), 8 waves/block, 16 queries/wave.
// K/V tiles for the block's shared (b,head) are staged once per 32-key tile
// via async global->LDS (ASYNCcnt); online softmax in registers; P relayout
// through a per-wave padded LDS tile; all matmuls via bf16 WMMA.
// ---------------------------------------------------------------------------
__global__ __launch_bounds__(256) void attn_wmma_kernel(
    const bf16* __restrict__ Q, const bf16* __restrict__ K,
    const bf16* __restrict__ Vt, bf16* __restrict__ AO)
{
    __shared__ bf16 sK[32 * 136];    // [32 keys][128 hd], 272B row stride
    __shared__ bf16 sV[128 * 40];    // [128 hd][32 keys], 80B row stride
    __shared__ bf16 sp[8][16][40];   // per-wave P staging (16 q x 32 keys)

    const int tid   = threadIdx.x;
    const int lane  = tid & 31;
    const int wave  = tid >> 5;
    const int laneN = lane & 15;
    const int hi    = lane >> 4;      // C/A frag half select
    const int bh    = blockIdx.y;
    const int b     = bh >> 4;
    const int head  = bh & 15;
    const int qblock = blockIdx.x * 128;
    const int qbase  = qblock + wave * 16;
    const float scale = 0.08838834764831845f;   // 1/sqrt(128)

    // Q fragments for this wave's 16 query rows: [16 x 128] = 4 A-frags
    v16bf aq[4];
    {
        const bf16* qp = Q + (size_t)(b * SS + qbase + laneN) * HH + head * HD + hi * 8;
        #pragma unroll
        for (int kc = 0; kc < 4; kc++) aq[kc] = ld_frag(qp + kc * 32);
    }

    v8f o[8];
    float mrun[8], lrun[8];
    #pragma unroll
    for (int t = 0; t < 8; t++) o[t] = (v8f){0.f, 0.f, 0.f, 0.f, 0.f, 0.f, 0.f, 0.f};
    #pragma unroll
    for (int g = 0; g < 8; g++) { mrun[g] = -1e30f; lrun[g] = 0.f; }

    const int blk_ktmax = (qblock + 127) >> 5;   // block-shared key-tile range
    const int my_ktmax  = (qbase + 15) >> 5;     // wave-uniform causal bound

    for (int kt = 0; kt <= blk_ktmax; kt++) {
        const int kb = kt << 5;

        // ---- async-stage shared K tile [32 x 128] and V tile [128 x 32] ----
        __syncthreads();
        #pragma unroll
        for (int i = 0; i < 2; i++) {
            const int cc  = tid + i * 256;        // 0..511
            const int krw = cc >> 4, kcl = cc & 15;
            async_copy_b128(sK + krw * 136 + kcl * 8,
                            K + (size_t)(b * SS + kb + krw) * HH + head * HD + kcl * 8);
        }
        #pragma unroll
        for (int i = 0; i < 2; i++) {
            const int cc  = tid + i * 256;        // 0..511
            const int vrw = cc >> 2, vcl = cc & 3;
            async_copy_b128(sV + vrw * 40 + vcl * 8,
                            Vt + (size_t)(bh * HD + vrw) * SS + kb + vcl * 8);
        }
        wait_async0();
        __syncthreads();

        if (kt > my_ktmax) continue;   // wave-uniform: EXEC stays all-ones

        // ---- scores: two 16x16 tiles, K-reduction over hd=128 (4 WMMAs each)
        v8f sc[2];
        #pragma unroll
        for (int sub = 0; sub < 2; sub++) {
            v8f s = (v8f){0.f, 0.f, 0.f, 0.f, 0.f, 0.f, 0.f, 0.f};
            const bf16* kp = sK + (sub * 16 + laneN) * 136 + hi * 8;
            #pragma unroll
            for (int kc = 0; kc < 4; kc++) {
                v16bf bk = ld_frag(kp + kc * 32);
                s = wmma_bf16(aq[kc], bk, s);
            }
            const int kg = kb + sub * 16 + laneN;
            #pragma unroll
            for (int g = 0; g < 8; g++) {
                const int qg = qbase + g + 8 * hi;
                const float v = s[g] * scale;
                s[g] = (kg > qg) ? -1e30f : v;
            }
            sc[sub] = s;
        }

        // ---- online softmax update (per C-frag row g) ----
        float alpha[8];
        #pragma unroll
        for (int g = 0; g < 8; g++) {
            float rm = fmaxf(sc[0][g], sc[1][g]);
            #pragma unroll
            for (int mm = 1; mm < 16; mm <<= 1) rm = fmaxf(rm, __shfl_xor(rm, mm, 32));
            const float mn = fmaxf(mrun[g], rm);
            alpha[g] = __expf(mrun[g] - mn);
            mrun[g]  = mn;
            const float p0 = __expf(sc[0][g] - mn);
            const float p1 = __expf(sc[1][g] - mn);
            sc[0][g] = p0; sc[1][g] = p1;
            float rs = p0 + p1;
            #pragma unroll
            for (int mm = 1; mm < 16; mm <<= 1) rs += __shfl_xor(rs, mm, 32);
            lrun[g] = lrun[g] * alpha[g] + rs;
        }
        #pragma unroll
        for (int t = 0; t < 8; t++)
            #pragma unroll
            for (int g = 0; g < 8; g++) o[t][g] *= alpha[g];

        // ---- P (C-frag layout) -> LDS -> A-frag layout ----
        #pragma unroll
        for (int sub = 0; sub < 2; sub++)
            #pragma unroll
            for (int g = 0; g < 8; g++)
                sp[wave][g + 8 * hi][sub * 16 + laneN] = (bf16)sc[sub][g];
        asm volatile("" ::: "memory");   // wave-private LDS; block compiler reorder
        v16bf ap = ld_frag(&sp[wave][laneN][hi * 8]);

        // ---- O += P @ V : 8 WMMAs over hd tiles (sV is [d][key], key-major) --
        #pragma unroll
        for (int t = 0; t < 8; t++) {
            v16bf bv = ld_frag(sV + (t * 16 + laneN) * 40 + hi * 8);
            o[t] = wmma_bf16(ap, bv, o[t]);
        }
    }

    // ---- finalize: O /= l, store bf16 attn-out [B*S, H] ----
    #pragma unroll
    for (int g = 0; g < 8; g++) {
        const float inv = 1.0f / lrun[g];
        const size_t gm = (size_t)(b * SS + qbase + g + 8 * hi);
        #pragma unroll
        for (int t = 0; t < 8; t++)
            AO[gm * HH + head * HD + t * 16 + laneN] = (bf16)(o[t][g] * inv);
    }
}

// ---------------------------------------------------------------------------
extern "C" void kernel_launch(void* const* d_in, const int* in_sizes, int n_in,
                              void* d_out, int out_size, void* d_ws, size_t ws_size,
                              hipStream_t stream)
{
    const float* h  = (const float*)d_in[0];
    const float* Wq = (const float*)d_in[1];
    const float* bq = (const float*)d_in[2];
    const float* Wk = (const float*)d_in[3];
    const float* bk = (const float*)d_in[4];
    const float* Wv = (const float*)d_in[5];
    const float* bv = (const float*)d_in[6];
    const float* Wo = (const float*)d_in[7];
    const float* bo = (const float*)d_in[8];
    float* out = (float*)d_out;

    const size_t szQ = (size_t)MTOT * HH * sizeof(bf16);   // 32 MiB each
    bf16* Qb  = (bf16*)d_ws;
    bf16* Kb  = (bf16*)((char*)d_ws + szQ);
    bf16* Vtb = (bf16*)((char*)d_ws + 2 * szQ);
    bf16* AO  = (bf16*)((char*)d_ws + 3 * szQ);            // total 128 MiB

    dim3 gg(MTOT / 64, HH / 256);
    gemm_wmma_kernel<0, false><<<gg, 256, 0, stream>>>(h, Wq, bq, Qb);
    gemm_wmma_kernel<0, false><<<gg, 256, 0, stream>>>(h, Wk, bk, Kb);
    gemm_wmma_kernel<1, false><<<gg, 256, 0, stream>>>(h, Wv, bv, Vtb);
    attn_wmma_kernel<<<dim3(SS / 128, BB * NHD), 256, 0, stream>>>(Qb, Kb, Vtb, AO);
    gemm_wmma_kernel<2, true><<<gg, 256, 0, stream>>>(AO, Wo, bo, out);
}
